// MobileNetV2_MINI_RESNET_4bitINP_89472758710751
// MI455X (gfx1250) — compile-verified, matrix operand order checked by reading, not run on hardware
//
#include <hip/hip_runtime.h>
#include <hip/hip_bf16.h>
#include <cstdint>

typedef __attribute__((ext_vector_type(8))) int v8i;
typedef __attribute__((ext_vector_type(2))) int v2i;

__device__ __forceinline__ int clampi(int v, int lo, int hi) {
    return v < lo ? lo : (v > hi ? hi : v);
}

// ---------------------------------------------------------------------------
// Input quantization: f32 NCHW (64,3,224,224) -> int4 codes stored as int8,
// NHWC layout.  code = clip(round((2x-1)/0.125), -8, 7)
// ---------------------------------------------------------------------------
__global__ void k_quant_input(const float* __restrict__ x,
                              int8_t* __restrict__ out, int total)
{
    int i = blockIdx.x * blockDim.x + threadIdx.x;
    if (i >= total) return;
    int c = i % 3; int t = i / 3;
    int xw = t % 224; t /= 224;
    int yh = t % 224; int n = t / 224;
    float v = x[((size_t)(n * 3 + c) * 224 + yh) * 224 + xw];
    v = 2.0f * v - 1.0f;
    out[i] = (int8_t)clampi((int)rintf(v * 8.0f), -8, 7);
}

// ---------------------------------------------------------------------------
// Per-output-channel weight quant (narrow int4) + BN fold.
//   s     = max|w_oc| / qmax + 1e-8
//   inv   = g / sqrt(v + 1e-5)
//   alpha = s * inv ;  beta = b - m * inv
// One block (64 thr) per output channel; grid is padded to a multiple of 16
// channels and padded rows are written as zeros (so WMMA B tiles never need
// bounds masking).
// ---------------------------------------------------------------------------
__global__ void k_prep_weights(const float* __restrict__ w,
                               const float* __restrict__ g, const float* __restrict__ b,
                               const float* __restrict__ m, const float* __restrict__ v,
                               int K, int qmax, int OC,
                               int8_t* __restrict__ wq,
                               float* __restrict__ alpha, float* __restrict__ beta)
{
    __shared__ float red[64];
    int oc = blockIdx.x, t = threadIdx.x;
    if (oc >= OC) {                      // zero padding row (block-uniform path)
        for (int i = t; i < K; i += 64) wq[(size_t)oc * K + i] = 0;
        if (t == 0) { alpha[oc] = 0.f; beta[oc] = 0.f; }
        return;
    }
    const float* wr = w + (size_t)oc * K;
    float mx = 0.f;
    for (int i = t; i < K; i += 64) mx = fmaxf(mx, fabsf(wr[i]));
    red[t] = mx; __syncthreads();
    for (int s = 32; s > 0; s >>= 1) {
        if (t < s) red[t] = fmaxf(red[t], red[t + s]);
        __syncthreads();
    }
    float sc = red[0] / (float)qmax + 1e-8f;
    for (int i = t; i < K; i += 64)
        wq[(size_t)oc * K + i] = (int8_t)clampi((int)rintf(wr[i] / sc), -qmax, qmax);
    if (t == 0) {
        float inv = g[oc] / sqrtf(v[oc] + 1e-5f);
        alpha[oc] = sc * inv;
        beta[oc]  = b[oc] - m[oc] * inv;
    }
}

// ---------------------------------------------------------------------------
// Stem: 3x3 stride-2 pad-1 conv, Cin=3, Cout=32, integer MACs, qrelu epilogue.
// ---------------------------------------------------------------------------
__global__ void k_stem_conv(const int8_t* __restrict__ a, const int8_t* __restrict__ wq,
                            const float* __restrict__ alpha, const float* __restrict__ beta,
                            const float* __restrict__ s_out_p,
                            int8_t* __restrict__ out, int total)
{
    int i = blockIdx.x * blockDim.x + threadIdx.x;
    if (i >= total) return;
    int oc = i & 31; int t = i >> 5;
    int x = t % 112; t /= 112;
    int y = t % 112; int n = t / 112;
    int acc = 0;
    const int8_t* wr = wq + oc * 27;
    #pragma unroll
    for (int kh = 0; kh < 3; kh++) {
        int iy = 2 * y + kh - 1;
        if ((unsigned)iy >= 224u) continue;
        #pragma unroll
        for (int kw = 0; kw < 3; kw++) {
            int ix = 2 * x + kw - 1;
            if ((unsigned)ix >= 224u) continue;
            const int8_t* ap = a + ((size_t)(n * 224 + iy) * 224 + ix) * 3;
            acc += (int)ap[0] * (int)wr[0 * 9 + kh * 3 + kw]
                 + (int)ap[1] * (int)wr[1 * 9 + kh * 3 + kw]
                 + (int)ap[2] * (int)wr[2 * 9 + kh * 3 + kw];
        }
    }
    float yf = (float)acc * (0.125f * alpha[oc]) + beta[oc];
    float inv_so = 1.0f / (*s_out_p);
    out[i] = (int8_t)clampi((int)rintf(fmaxf(yf, 0.f) * inv_so), 0, 15);
}

// ---------------------------------------------------------------------------
// Depthwise 3x3 conv (stride 1/2, pad 1), integer MACs, qrelu epilogue.
// ---------------------------------------------------------------------------
__global__ void k_dw_conv(const int8_t* __restrict__ a, const int8_t* __restrict__ wq,
                          const float* __restrict__ alpha, const float* __restrict__ beta,
                          const float* __restrict__ s_in_p, const float* __restrict__ s_out_p,
                          int8_t* __restrict__ out,
                          int C, int Hin, int Hout, int stride, int total)
{
    int i = blockIdx.x * blockDim.x + threadIdx.x;
    if (i >= total) return;
    int c = i % C; int t = i / C;
    int x = t % Hout; t /= Hout;
    int y = t % Hout; int n = t / Hout;
    const int8_t* wr = wq + c * 9;
    int acc = 0;
    #pragma unroll
    for (int kh = 0; kh < 3; kh++) {
        int iy = y * stride + kh - 1;
        if ((unsigned)iy >= (unsigned)Hin) continue;
        #pragma unroll
        for (int kw = 0; kw < 3; kw++) {
            int ix = x * stride + kw - 1;
            if ((unsigned)ix >= (unsigned)Hin) continue;
            acc += (int)a[((size_t)(n * Hin + iy) * Hin + ix) * C + c] * (int)wr[kh * 3 + kw];
        }
    }
    float yf = (float)acc * ((*s_in_p) * alpha[c]) + beta[c];
    float inv_so = 1.0f / (*s_out_p);
    out[i] = (int8_t)clampi((int)rintf(fmaxf(yf, 0.f) * inv_so), 0, 15);
}

// ---------------------------------------------------------------------------
// Pointwise conv as integer GEMM via V_WMMA_I32_16X16X64_IU8, CIN known at
// compile time (multiple of 8, <=128).
//   A: [M, CIN] int8 activation codes (SGN_A: signed residual-path / unsigned
//      post-relu codes)
//   B: [Cout_padded16, CIN] int8 signed weight codes, zero-padded rows.
// Per-lane fragment layout (ISA 7.12.2, 8-bit A/B): VGPR v holds 4 K-bytes at
// K = (v>>1)*16 + half*8 + (v&1)*4, so dword pairs form contiguous 8-byte
// chunks at k0 + j*16 + half*8 -> four global_load_b64 per matrix per K-step.
// Epilogue modes: 0 = qrelu, 1 = qint, 2 = quantized residual add.
// Quantization divisors hoisted to per-thread reciprocals (1-2 divisions per
// thread instead of 8-16 IEEE division chains).
// ---------------------------------------------------------------------------
template <int CIN, bool SGN_A>
__global__ void k_pw_wmma(const int8_t* __restrict__ A, const int8_t* __restrict__ Wq,
                          const float* __restrict__ alpha, const float* __restrict__ beta,
                          const float* __restrict__ s_in_p, const float* __restrict__ s_out_p,
                          const float* __restrict__ shared_p, const int8_t* __restrict__ res,
                          int8_t* __restrict__ out, int Cout, int mode)
{
    int lane = threadIdx.x & 31;
    int wave = threadIdx.x >> 5;
    int r = lane & 15;          // A row within tile / B column within tile
    int h = (lane >> 4) & 1;    // lane half (K interleave)
    int m0 = (blockIdx.x * 8 + wave) * 16;
    int n  = blockIdx.y * 16 + r;

    const int8_t* arow = A  + (size_t)(m0 + r) * CIN;
    const int8_t* brow = Wq + (size_t)n * CIN;   // rows padded to mult. of 16

    v8i acc = {};
    #pragma unroll
    for (int k0 = 0; k0 < CIN; k0 += 64) {
        v8i av = {}, bv = {};
        #pragma unroll
        for (int j = 0; j < 4; j++) {
            const int kbase = k0 + j * 16;       // compile-time constant
            v2i a2 = {0, 0}, b2 = {0, 0};
            if (kbase + 16 <= CIN) {
                // both lane-halves in range: unconditional b64 loads
                a2 = *(const v2i*)(arow + kbase + h * 8);
                b2 = *(const v2i*)(brow + kbase + h * 8);
            } else if (kbase + 8 <= CIN) {
                // only CIN%16==8 tail: low half valid; load valid address for
                // all lanes, branchlessly zero the h==1 half
                v2i at = *(const v2i*)(arow + kbase);
                v2i bt = *(const v2i*)(brow + kbase);
                a2.x = h ? 0 : at.x;  a2.y = h ? 0 : at.y;
                b2.x = h ? 0 : bt.x;  b2.y = h ? 0 : bt.y;
            }
            av[2 * j] = a2.x;  av[2 * j + 1] = a2.y;
            bv[2 * j] = b2.x;  bv[2 * j + 1] = b2.y;
        }
        if (SGN_A)
            acc = __builtin_amdgcn_wmma_i32_16x16x64_iu8(true,  av, true, bv, acc, false, false);
        else
            acc = __builtin_amdgcn_wmma_i32_16x16x64_iu8(false, av, true, bv, acc, false, false);
    }

    if (n >= Cout) return;
    float sin = *s_in_p;
    float so  = *s_out_p;
    float al  = sin * alpha[n];
    float be  = beta[n];
    float inv_so = 1.0f / so;            // single division per thread

    if (mode == 0) {                    // qrelu -> uint4 codes
        #pragma unroll
        for (int g = 0; g < 8; g++) {
            int mrow = m0 + g + h * 8;
            float yf = (float)acc[g] * al + be;
            out[(size_t)mrow * Cout + n] =
                (int8_t)clampi((int)rintf(fmaxf(yf, 0.f) * inv_so), 0, 15);
        }
    } else if (mode == 1) {             // qint -> int4 codes
        #pragma unroll
        for (int g = 0; g < 8; g++) {
            int mrow = m0 + g + h * 8;
            float yf = (float)acc[g] * al + be;
            out[(size_t)mrow * Cout + n] =
                (int8_t)clampi((int)rintf(yf * inv_so), -8, 7);
        }
    } else {                            // quantized residual add
        float sh = *shared_p;
        float inv_sh = 1.0f / sh;        // second (and last) division
        float ratio  = sh * inv_so;      // sh / so
        #pragma unroll
        for (int g = 0; g < 8; g++) {
            int mrow = m0 + g + h * 8;
            float yf = (float)acc[g] * al + be;
            int iy = clampi((int)rintf(yf * inv_sh), -8, 7);
            int xv = (int)res[(size_t)mrow * Cout + n];
            out[(size_t)mrow * Cout + n] =
                (int8_t)clampi((int)rintf((float)(xv + iy) * ratio), -8, 7);
        }
    }
}

// ---------------------------------------------------------------------------
// TruncAvgPool(14) + 8-bit QuantLinear.  grid=64 (batch), block=128 (channel).
// ---------------------------------------------------------------------------
__global__ void k_pool_fc(const int8_t* __restrict__ acts, const float* __restrict__ s_head_p,
                          const float* __restrict__ Wfc, const float* __restrict__ bfc,
                          float* __restrict__ out)
{
    __shared__ float smax[128];
    __shared__ float sum0[128];
    __shared__ float sum1[128];
    int b = blockIdx.x, c = threadIdx.x;

    int acc = 0;
    const int8_t* base = acts + (size_t)b * 196 * 128 + c;
    for (int i = 0; i < 196; i++) acc += (int)base[i * 128];
    int q = acc >> 4;                                    // floor(acc / 2^4), acc >= 0
    float shd = *s_head_p;
    float in_scale = shd * 16.0f / 196.0f;
    float pooled = (float)q * in_scale;

    float w0 = Wfc[c], w1 = Wfc[128 + c];
    smax[c] = fmaxf(fabsf(w0), fabsf(w1));
    __syncthreads();
    for (int s = 64; s > 0; s >>= 1) {
        if (c < s) smax[c] = fmaxf(smax[c], smax[c + s]);
        __syncthreads();
    }
    float sw = smax[0] / 127.0f + 1e-8f;
    float inv_sw = 1.0f / sw;
    float wq0 = (float)clampi((int)rintf(w0 * inv_sw), -127, 127) * sw;
    float wq1 = (float)clampi((int)rintf(w1 * inv_sw), -127, 127) * sw;
    sum0[c] = pooled * wq0;
    sum1[c] = pooled * wq1;
    __syncthreads();
    for (int s = 64; s > 0; s >>= 1) {
        if (c < s) { sum0[c] += sum0[c + s]; sum1[c] += sum1[c + s]; }
        __syncthreads();
    }
    if (c == 0) {
        float bs = in_scale * sw;
        float inv_bs = 1.0f / bs;
        out[b * 2 + 0] = sum0[0] + rintf(bfc[0] * inv_bs) * bs;
        out[b * 2 + 1] = sum1[0] + rintf(bfc[1] * inv_bs) * bs;
    }
}

// ---------------------------------------------------------------------------
// Host-side dispatch over compile-time CIN
// ---------------------------------------------------------------------------
struct PwArgs {
    const int8_t* A; const int8_t* Wq; const float* alpha; const float* beta;
    const float* s_in; const float* s_out; const float* shared; const int8_t* res;
    int8_t* out; int M; int Cout; int mode;
};

template <bool SGN>
static void launch_pw(int Cin, const PwArgs& a, hipStream_t stream)
{
    dim3 grid(a.M / 128, (a.Cout + 15) / 16);
    dim3 blk(256);
#define PW_CASE(C) case C: \
    k_pw_wmma<C, SGN><<<grid, blk, 0, stream>>>( \
        a.A, a.Wq, a.alpha, a.beta, a.s_in, a.s_out, a.shared, a.res, \
        a.out, a.Cout, a.mode); break;
    switch (Cin) {
        PW_CASE(8) PW_CASE(16) PW_CASE(24) PW_CASE(32)
        PW_CASE(48) PW_CASE(64) PW_CASE(96) PW_CASE(128)
        default: break;
    }
#undef PW_CASE
}

// ---------------------------------------------------------------------------
// Host orchestration
// ---------------------------------------------------------------------------
extern "C" void kernel_launch(void* const* d_in, const int* in_sizes, int n_in,
                              void* d_out, int out_size, void* d_ws, size_t ws_size,
                              hipStream_t stream)
{
    (void)n_in; (void)out_size; (void)ws_size;

    struct BIdx { int w_pw1, bn1[4], s1, w_dw, bn2[4], s2, w_pw2, bn3[4], s_out; };
    int Ix, Iw_stem, Ibn_stem[4], Is_stem, Iw_head, Ibn_head[4], Is_head, Iw_fc, Ib_fc;
    BIdx B[10];

    // Detect flattening order of the nested input dict:
    //   insertion order -> d_in[0] is x (9,633,792 elems)
    //   jax tree_leaves (sorted keys) -> d_in[0] is b_fc (2 elems)
    bool insertion = in_sizes[0] > 1000;
    if (insertion) {
        Ix = 0; Iw_stem = 1;
        for (int j = 0; j < 4; j++) Ibn_stem[j] = 2 + j;   // g,b,m,v
        Is_stem = 6;
        B[0].w_pw1 = -1; B[0].w_dw = 7;
        for (int j = 0; j < 4; j++) B[0].bn2[j] = 8 + j;
        B[0].s2 = 12; B[0].w_pw2 = 13;
        for (int j = 0; j < 4; j++) B[0].bn3[j] = 14 + j;
        B[0].s_out = 18;
        for (int b = 1; b < 10; b++) {
            int base = 19 + (b - 1) * 18;
            B[b].w_pw1 = base;
            for (int j = 0; j < 4; j++) B[b].bn1[j] = base + 1 + j;
            B[b].s1 = base + 5; B[b].w_dw = base + 6;
            for (int j = 0; j < 4; j++) B[b].bn2[j] = base + 7 + j;
            B[b].s2 = base + 11; B[b].w_pw2 = base + 12;
            for (int j = 0; j < 4; j++) B[b].bn3[j] = base + 13 + j;
            B[b].s_out = base + 17;
        }
        Iw_head = 181;
        for (int j = 0; j < 4; j++) Ibn_head[j] = 182 + j;
        Is_head = 186; Iw_fc = 187; Ib_fc = 188;
    } else {
        // sorted keys; bn dicts flatten b,g,m,v -> map to (g,b,m,v)
        Ib_fc = 0;
        B[0].w_pw1 = -1;
        B[0].bn2[0] = 2;  B[0].bn2[1] = 1;  B[0].bn2[2] = 3;  B[0].bn2[3] = 4;
        B[0].bn3[0] = 6;  B[0].bn3[1] = 5;  B[0].bn3[2] = 7;  B[0].bn3[3] = 8;
        B[0].s2 = 9; B[0].s_out = 10; B[0].w_dw = 11; B[0].w_pw2 = 12;
        for (int b = 1; b < 10; b++) {
            int base = 13 + (b - 1) * 18;
            B[b].bn1[0] = base + 1;  B[b].bn1[1] = base + 0;  B[b].bn1[2] = base + 2;  B[b].bn1[3] = base + 3;
            B[b].bn2[0] = base + 5;  B[b].bn2[1] = base + 4;  B[b].bn2[2] = base + 6;  B[b].bn2[3] = base + 7;
            B[b].bn3[0] = base + 9;  B[b].bn3[1] = base + 8;  B[b].bn3[2] = base + 10; B[b].bn3[3] = base + 11;
            B[b].s1 = base + 12; B[b].s2 = base + 13; B[b].s_out = base + 14;
            B[b].w_dw = base + 15; B[b].w_pw1 = base + 16; B[b].w_pw2 = base + 17;
        }
        int base2 = 13 + 9 * 18; // 175
        Ibn_head[0] = base2 + 1; Ibn_head[1] = base2 + 0; Ibn_head[2] = base2 + 2; Ibn_head[3] = base2 + 3;
        Ibn_stem[0] = base2 + 5; Ibn_stem[1] = base2 + 4; Ibn_stem[2] = base2 + 6; Ibn_stem[3] = base2 + 7;
        Is_head = base2 + 8; Is_stem = base2 + 9;
        Iw_fc = base2 + 10; Iw_head = base2 + 11; Iw_stem = base2 + 12;
        Ix = base2 + 13;
    }

    auto P = [&](int i) -> const float* { return (const float*)d_in[i]; };

    // ---- workspace carving ----
    char* wsp = (char*)d_ws;
    size_t off = 0;
    auto alloc = [&](size_t bytes) -> char* {
        char* p = wsp + off;
        off += (bytes + 255) & ~(size_t)255;
        return p;
    };
    int8_t* in_codes = (int8_t*)alloc(64ull * 224 * 224 * 3);
    int8_t* bufs[4];
    for (int i = 0; i < 4; i++) bufs[i] = (int8_t*)alloc(64ull * 112 * 112 * 32);

    struct Layer { int8_t* wq; float* alpha; float* beta; };
    auto prep = [&](int wIdx, const int* bn, int OC, int K) -> Layer {
        int OCpad = (OC + 15) & ~15;
        Layer L;
        L.wq    = (int8_t*)alloc((size_t)OCpad * K);
        L.alpha = (float*)alloc((size_t)OCpad * 4);
        L.beta  = (float*)alloc((size_t)OCpad * 4);
        k_prep_weights<<<dim3(OCpad), dim3(64), 0, stream>>>(
            P(wIdx), P(bn[0]), P(bn[1]), P(bn[2]), P(bn[3]), K, 7, OC,
            L.wq, L.alpha, L.beta);
        return L;
    };

    // ---- input quant ----
    {
        int total = 64 * 224 * 224 * 3;
        k_quant_input<<<dim3((total + 255) / 256), dim3(256), 0, stream>>>(
            P(Ix), in_codes, total);
    }

    // ---- stem ----
    Layer Lstem = prep(Iw_stem, Ibn_stem, 32, 27);
    int8_t* xbuf = bufs[0];
    int8_t* outbuf = bufs[3];
    int8_t* tmp1 = bufs[1];
    int8_t* tmp2 = bufs[2];
    {
        int total = 64 * 112 * 112 * 32;
        k_stem_conv<<<dim3((total + 255) / 256), dim3(256), 0, stream>>>(
            in_codes, Lstem.wq, Lstem.alpha, Lstem.beta, P(Is_stem), xbuf, total);
    }

    // ---- inverted residual blocks ----
    const int binp[10] = {32, 8, 16, 16, 24, 24, 32, 32, 32, 64};
    const int boup[10] = {8, 16, 16, 24, 24, 32, 32, 32, 64, 64};
    const int bst[10]  = {1, 2, 1, 2, 1, 2, 1, 1, 1, 1};
    const int bt[10]   = {1, 2, 2, 2, 2, 4, 4, 4, 2, 2};
    const int bid[10]  = {0, 0, 1, 0, 1, 0, 1, 1, 0, 1};

    const float* prev_sout = P(Is_stem);
    int H = 112;
    for (int b = 0; b < 10; b++) {
        int inp = binp[b], oup = boup[b], st = bst[b], t = bt[b];
        int hch = inp * t;
        int Hout = (st == 2) ? H / 2 : H;
        const int8_t* cur = xbuf;
        const float* cur_scale = prev_sout;

        if (t != 1) {
            Layer L1 = prep(B[b].w_pw1, B[b].bn1, hch, inp);
            PwArgs a{xbuf, L1.wq, L1.alpha, L1.beta, prev_sout, P(B[b].s1),
                     nullptr, nullptr, tmp1, 64 * H * H, hch, 0};
            launch_pw<true>(inp, a, stream);          // signed int4 activations
            cur = tmp1;
            cur_scale = P(B[b].s1);
        }

        Layer Ld = prep(B[b].w_dw, B[b].bn2, hch, 9);
        {
            int total = 64 * Hout * Hout * hch;
            k_dw_conv<<<dim3((total + 255) / 256), dim3(256), 0, stream>>>(
                cur, Ld.wq, Ld.alpha, Ld.beta, cur_scale, P(B[b].s2),
                tmp2, hch, H, Hout, st, total);
        }

        Layer L2 = prep(B[b].w_pw2, B[b].bn3, oup, hch);
        {
            PwArgs a{tmp2, L2.wq, L2.alpha, L2.beta, P(B[b].s2), P(B[b].s_out),
                     prev_sout, xbuf, outbuf, 64 * Hout * Hout, oup,
                     bid[b] ? 2 : 1};
            launch_pw<false>(hch, a, stream);         // unsigned uint4 activations
        }

        int8_t* t0 = xbuf; xbuf = outbuf; outbuf = t0;
        prev_sout = P(B[b].s_out);
        H = Hout;
    }

    // ---- head 1x1 conv (64 -> 128, qrelu) ----
    Layer Lh = prep(Iw_head, Ibn_head, 128, 64);
    {
        PwArgs a{xbuf, Lh.wq, Lh.alpha, Lh.beta, prev_sout, P(Is_head),
                 nullptr, nullptr, tmp1, 64 * 14 * 14, 128, 0};
        launch_pw<true>(64, a, stream);
    }

    // ---- trunc-avg-pool + quantized FC ----
    k_pool_fc<<<dim3(64), dim3(128), 0, stream>>>(
        tmp1, P(Is_head), P(Iw_fc), P(Ib_fc), (float*)d_out);
}
